// DeformableAttnBlock_FUSION_69355131896290
// MI455X (gfx1250) — compile-verified, matrix operand order checked by reading, not run on hardware
//
#include <hip/hip_runtime.h>

#define HH 256
#define WW 256
#define HWX (HH * WW)

typedef __attribute__((ext_vector_type(16))) __bf16       v16bf;
typedef __attribute__((ext_vector_type(8)))  __bf16       v8bf;
typedef __attribute__((ext_vector_type(8)))  float        v8f;
typedef __attribute__((ext_vector_type(4)))  unsigned int v4u;
typedef __attribute__((ext_vector_type(8)))  int          v8i;
typedef __attribute__((ext_vector_type(4)))  int          v4i;

#if defined(__has_builtin)
#  if __has_builtin(__builtin_amdgcn_tensor_load_to_lds) && \
      __has_builtin(__builtin_amdgcn_s_wait_tensorcnt)
#    define USE_TDM 1
#  endif
#endif
#ifndef USE_TDM
#  define USE_TDM 0
#endif

// ---------------------------------------------------------------------------
// Weight repack: f32 [Cout][Cin][3][3] -> bf16 [Cout][taps][Cin_pad] (r-major K)
// ---------------------------------------------------------------------------
__global__ __launch_bounds__(256)
void repack_w_k(const float* __restrict__ src, __bf16* __restrict__ dst,
                int Cout, int Cin, int taps, int Cpad)
{
  int idx = blockIdx.x * 256 + threadIdx.x;
  const int total = Cout * taps * Cpad;
  if (idx >= total) return;
  const int co  = idx / (taps * Cpad);
  const int rem = idx - co * taps * Cpad;
  const int r   = rem / Cpad;
  const int ci  = rem - r * Cpad;
  float v = (ci < Cin) ? src[((size_t)co * Cin + ci) * taps + r] : 0.f;
  dst[idx] = (__bf16)v;
}

// ---------------------------------------------------------------------------
// f32 NCHW -> bf16 NHWC (channel padded)
// ---------------------------------------------------------------------------
__global__ __launch_bounds__(256)
void nchw_to_nhwc_k(const float* __restrict__ src, __bf16* __restrict__ dst,
                    int nimg, int C, int Cpad)
{
  const int HW = HWX;
  int idx = blockIdx.x * 256 + threadIdx.x;
  if (idx >= nimg * HW) return;
  const int n = idx / HW, p = idx - n * HW;
  __bf16* d = dst + ((size_t)n * HW + p) * Cpad;
  for (int c = 0; c < Cpad; ++c)
    d[c] = (c < C) ? (__bf16)src[((size_t)n * C + c) * HW + p] : (__bf16)0.f;
}

// ---------------------------------------------------------------------------
// Implicit-GEMM conv on WMMA bf16 16x16x32, f32 accumulate.
// Activations: bf16 NHWC [n][HW][Cpad], Cpad % 32 == 0.
// Weights:     bf16 [Cout][taps][Cpad]; 16-row tile staged in LDS via TDM.
// One wave -> 32 pixels (2 M-tiles) x 16 Cout; 2 WMMAs share each B fragment.
// grid = (HW/256, Cout/16, nimg), block 256 (8 waves).
// outMode 0: bf16 NHWC with outC channel split; outMode 1: f32 NCHW.
// resMode  0: none; 1: f32 NCHW (+resStride per image); 2: bf16 NHWC (Cout ch).
// ---------------------------------------------------------------------------
__global__ __launch_bounds__(256)
void conv_wmma_k(const __bf16* __restrict__ in, const __bf16* __restrict__ wgt,
                 const float* __restrict__ bias, void* __restrict__ out,
                 const void* __restrict__ res, long long resStride,
                 int Cpad, int Cout, int taps, int dil, int act,
                 int outMode, int outC, int resMode)
{
  const int H = HH, W = WW, HW = HWX;
  const int Ktot = taps * Cpad;
  const int lane = threadIdx.x & 31;
  const int wave = threadIdx.x >> 5;
  const int row  = lane & 15;        // A: pixel (M).  B: cout (N).
  const int kh   = lane >> 4;        // K-half per ISA 16-bit fragment layout
  const int khoff = kh << 3;

  const int p0  = (blockIdx.x * 8 + wave) * 32;
  const int co0 = blockIdx.y * 16;
  const int n   = blockIdx.z;

  // ---- stage B tile (16 cout rows x Ktot bf16, contiguous) into LDS --------
  extern __shared__ char smem_raw[];
  __bf16* wlds = (__bf16*)smem_raw;
#if USE_TDM
  if (wave == 0) {
    // Tensor DMA descriptor: 1-D tile of 16*Ktot bf16 elements, global->LDS.
    const unsigned long long gaddr =
        (unsigned long long)(const void*)(wgt + (size_t)co0 * Ktot);
    const unsigned int ldsaddr = (unsigned int)(unsigned long long)(void*)wlds;
    const unsigned int nelem = 16u * (unsigned int)Ktot;   // <= 18432, fits 16b
    v4u g0;
    g0[0] = 1u;                                            // count=1, user D#
    g0[1] = ldsaddr;                                       // lds_addr (bytes)
    g0[2] = (unsigned int)(gaddr & 0xFFFFFFFFull);         // global_addr lo
    g0[3] = (unsigned int)((gaddr >> 32) & 0x01FFFFFFull)  // global_addr hi
          | (2u << 30);                                    // type = 2 (image)
    v8i g1;
    g1[0] = (int)(1u << 16);                  // data_size = 1 -> 2 bytes
    g1[1] = (int)((nelem & 0xFFFFu) << 16);   // tensor_dim0[15:0]
    g1[2] = (int)((nelem >> 16) | (1u << 16)); // tensor_dim0[31:16], dim1 = 1
    g1[3] = (int)((nelem & 0xFFFFu) << 16);   // tile_dim0 = nelem
    g1[4] = 0;                                // tile_dim1/2 unused
    g1[5] = (int)nelem;                       // tensor_dim0_stride lo
    g1[6] = 0;
    g1[7] = 0;
    v4i z4 = {};
#if defined(__clang_major__) && (__clang_major__ >= 23)
    v8i z8 = {};
    __builtin_amdgcn_tensor_load_to_lds(g0, g1, z4, z4, z8, 0);
#else
    __builtin_amdgcn_tensor_load_to_lds(g0, g1, z4, z4, 0);
#endif
    __builtin_amdgcn_s_wait_tensorcnt(0);
  }
#else
  {
    const uint4* s = (const uint4*)(wgt + (size_t)co0 * Ktot);
    uint4*       d = (uint4*)wlds;
    const int nvec = (16 * Ktot) >> 3;
    for (int i = threadIdx.x; i < nvec; i += 256) d[i] = s[i];
  }
#endif
  __syncthreads();

  const __bf16* __restrict__ wl    = wlds + row * Ktot;
  const __bf16* __restrict__ inImg = in + (size_t)n * HW * Cpad;

  const int pA = p0 + row;       // first tile's pixel for this lane
  const int py = pA / W;
  const int px = pA - py * W;    // second tile: px + 16, same py (32 | p0)

  v8f acc0 = {}, acc1 = {};
  for (int r = 0; r < taps; ++r) {
    const int dy2 = (taps == 9) ? (r / 3 - 1) * dil : 0;
    const int dx2 = (taps == 9) ? (r % 3 - 1) * dil : 0;
    const int yy = py + dy2;
    const bool yok = (yy >= 0 && yy < H);
    const int xx0 = px + dx2, xx1 = xx0 + 16;
    const bool ok0 = yok && (xx0 >= 0 && xx0 < W);
    const bool ok1 = yok && (xx1 >= 0 && xx1 < W);
    const __bf16* ap0 = inImg + (size_t)(yy * W + xx0) * Cpad;
    const __bf16* ap1 = inImg + (size_t)(yy * W + xx1) * Cpad;
    const int kbase = r * Cpad;
    for (int ks = 0; ks < Cpad; ks += 32) {
      const v8bf b0 = *(const v8bf*)(wl + kbase + ks + khoff);       // ds_load_b128
      const v8bf b1 = *(const v8bf*)(wl + kbase + ks + 16 + khoff);  // ds_load_b128
      v8bf a00 = {}, a01 = {}, a10 = {}, a11 = {};
      if (ok0) {
        a00 = *(const v8bf*)(ap0 + ks + khoff);        // global_load_b128
        a01 = *(const v8bf*)(ap0 + ks + 16 + khoff);
      }
      if (ok1) {
        a10 = *(const v8bf*)(ap1 + ks + khoff);
        a11 = *(const v8bf*)(ap1 + ks + 16 + khoff);
      }
      v16bf bm, aA, aB;
#pragma unroll
      for (int i = 0; i < 8; ++i) {
        bm[i] = b0[i];  bm[8 + i] = b1[i];
        aA[i] = a00[i]; aA[8 + i] = a01[i];
        aB[i] = a10[i]; aB[8 + i] = a11[i];
      }
      acc0 = __builtin_amdgcn_wmma_f32_16x16x32_bf16(
          false, aA, false, bm, (short)0, acc0, false, false);
      acc1 = __builtin_amdgcn_wmma_f32_16x16x32_bf16(
          false, aB, false, bm, (short)0, acc1, false, false);
    }
  }

  // ---- epilogue: C/D layout lane&15 = N (cout), VGPR rr = M (pixel, +8 hi) --
  const int co = co0 + row;
  const float bco = bias[co];
#pragma unroll
  for (int half = 0; half < 2; ++half) {
    const v8f acc = half ? acc1 : acc0;
    const int pbase = p0 + half * 16;
#pragma unroll
    for (int rr = 0; rr < 8; ++rr) {
      const int pm = pbase + rr + khoff;
      float v = acc[rr] + bco;
      if (resMode == 1)
        v += ((const float*)res)[(size_t)n * resStride + (size_t)co * HW + pm];
      else if (resMode == 2)
        v += (float)((const __bf16*)res)[((size_t)n * HW + pm) * Cout + co];
      if (act == 1) v = (v >= 0.0f) ? v : 0.1f * v;
      if (outMode == 0) {
        const int img = n * (Cout / outC) + co / outC;
        const int ch  = co % outC;
        ((__bf16*)out)[((size_t)img * HW + pm) * outC + ch] = (__bf16)v;
      } else {
        ((float*)out)[((size_t)n * Cout + co) * HW + pm] = v;
      }
    }
  }
}

// ---------------------------------------------------------------------------
// Flow-warp both neighbors + assemble qk_in, bf16 NHWC [b][p][128]:
// [warp01(32) | frame1(32) | warp21(32) | ff(2) | fb(2) | zero-pad(28)]
// ---------------------------------------------------------------------------
__global__ __launch_bounds__(256)
void warp_build_qkin_k(const float* __restrict__ frame, const float* __restrict__ ffwd,
                       const float* __restrict__ fbwd, __bf16* __restrict__ qk)
{
  const int HW = HWX, W = WW, H = HH, C = 32, T = 3;
  int idx = blockIdx.x * 256 + threadIdx.x;
  if (idx >= 2 * HW) return;
  const int b = idx / HW, p = idx - b * HW;
  const int y = p / W, x = p - y * W;

  const float* f0 = frame + (size_t)(b * T + 0) * C * HW;
  const float* f1 = frame + (size_t)(b * T + 1) * C * HW;
  const float* f2 = frame + (size_t)(b * T + 2) * C * HW;
  const float* fb = fbwd + (size_t)b * (T - 1) * 2 * HW;           // t = 0
  const float* ff = ffwd + ((size_t)b * (T - 1) + 1) * 2 * HW;     // t = 1
  __bf16* q = qk + ((size_t)b * HW + p) * 128;

  const float fbx = fb[p], fby = fb[HW + p];
  const float ffx = ff[p], ffy = ff[HW + p];

  auto warp = [&](const float* img, float fx, float fy, int chBase) {
    const float xs = (float)x + fx, ys = (float)y + fy;
    const float x0f = floorf(xs), y0f = floorf(ys);
    const int x0 = (int)x0f, y0 = (int)y0f;
    const float lx = xs - x0f, ly = ys - y0f;
    int ii[4]; float wt[4];
#pragma unroll
    for (int t = 0; t < 4; ++t) {
      const int dx = t & 1, dy = t >> 1;
      const int xi = x0 + dx, yi = y0 + dy;
      const bool vld = (xi >= 0 && xi < W && yi >= 0 && yi < H);
      const int xc = xi < 0 ? 0 : (xi > W - 1 ? W - 1 : xi);
      const int yc = yi < 0 ? 0 : (yi > H - 1 ? H - 1 : yi);
      ii[t] = yc * W + xc;
      wt[t] = vld ? (dx ? lx : 1.f - lx) * (dy ? ly : 1.f - ly) : 0.f;
    }
    for (int c = 0; c < C; ++c) {
      const float* ic = img + (size_t)c * HW;
      q[chBase + c] = (__bf16)(wt[0] * ic[ii[0]] + wt[1] * ic[ii[1]] +
                               wt[2] * ic[ii[2]] + wt[3] * ic[ii[3]]);
    }
  };

  warp(f0, fbx, fby, 0);
  for (int c = 0; c < C; ++c) q[32 + c] = (__bf16)f1[(size_t)c * HW + p];
  warp(f2, ffx, ffy, 64);
  q[96] = (__bf16)ffx; q[97] = (__bf16)ffy;
  q[98] = (__bf16)fbx; q[99] = (__bf16)fby;
  for (int c = 100; c < 128; ++c) q[c] = (__bf16)0.f;
}

// ---------------------------------------------------------------------------
// Softmax over 12 contiguous bf16 per (b, pixel, head); aw NHWC [b][p][48].
// ---------------------------------------------------------------------------
__global__ __launch_bounds__(256)
void softmax_aw_k(__bf16* __restrict__ aw)
{
  const int HW = HWX;
  int idx = blockIdx.x * 256 + threadIdx.x;
  if (idx >= 2 * 4 * HW) return;
  const int b = idx / (4 * HW);
  const int rem = idx - b * 4 * HW;
  const int hh = rem / HW, p = rem - hh * HW;
  __bf16* base = aw + ((size_t)b * HW + p) * 48 + hh * 12;
  float v[12], m = -3.4e38f;
#pragma unroll
  for (int i = 0; i < 12; ++i) { v[i] = (float)base[i]; m = fmaxf(m, v[i]); }
  float s = 0.f;
#pragma unroll
  for (int i = 0; i < 12; ++i) { v[i] = __expf(v[i] - m); s += v[i]; }
  const float inv = 1.f / s;
#pragma unroll
  for (int i = 0; i < 12; ++i) base[i] = (__bf16)(v[i] * inv);
}

// ---------------------------------------------------------------------------
// Multi-scale deformable attention, NHWC bf16. Sampling pos = (x+offx, y+offy).
// val: (6, HW, 32); off: (b, HW, 96); aw: (b, HW, 48); out: (b, HW, 32).
// ---------------------------------------------------------------------------
__global__ __launch_bounds__(256)
void deform_attn_k(const __bf16* __restrict__ val, const __bf16* __restrict__ off,
                   const __bf16* __restrict__ aw, __bf16* __restrict__ out)
{
  const int HW = HWX, W = WW, H = HH;
  int idx = blockIdx.x * 256 + threadIdx.x;
  if (idx >= 2 * 4 * HW) return;
  const int b = idx / (4 * HW);
  const int rem = idx - b * 4 * HW;
  const int hh = rem / HW, p = rem - hh * HW;
  const int y = p / W, x = p - y * W;

  const __bf16* offp = off + ((size_t)b * HW + p) * 96 + hh * 24;
  const __bf16* awp  = aw  + ((size_t)b * HW + p) * 48 + hh * 12;

  float acc[8] = {0.f, 0.f, 0.f, 0.f, 0.f, 0.f, 0.f, 0.f};

  for (int l = 0; l < 3; ++l) {
    const __bf16* vimg = val + (size_t)(b * 3 + l) * HW * 32;
    for (int pt = 0; pt < 4; ++pt) {
      const int lp = l * 4 + pt;
      const float ox = (float)offp[lp * 2];
      const float oy = (float)offp[lp * 2 + 1];
      const float wA = (float)awp[lp];
      const float xs = (float)x + ox, ys = (float)y + oy;
      const float x0f = floorf(xs), y0f = floorf(ys);
      const int x0 = (int)x0f, y0 = (int)y0f;
      const float lx = xs - x0f, ly = ys - y0f;
#pragma unroll
      for (int t = 0; t < 4; ++t) {
        const int dx = t & 1, dy = t >> 1;
        const int xi = x0 + dx, yi = y0 + dy;
        if (xi >= 0 && xi < W && yi >= 0 && yi < H) {
          const float wt = (dx ? lx : 1.f - lx) * (dy ? ly : 1.f - ly) * wA;
          const v8bf vv = *(const v8bf*)(vimg + ((size_t)(yi * W + xi) * 32 + hh * 8));
#pragma unroll
          for (int dd = 0; dd < 8; ++dd) acc[dd] += wt * (float)vv[dd];
        }
      }
    }
  }
  __bf16* op = out + ((size_t)b * HW + p) * 32 + hh * 8;
#pragma unroll
  for (int dd = 0; dd < 8; ++dd) op[dd] = (__bf16)acc[dd];
}

// ---------------------------------------------------------------------------
// tseq NHWC [b][p][64] = concat(out2 NHWC bf16, srcframe[:,1] f32 NCHW)
// ---------------------------------------------------------------------------
__global__ __launch_bounds__(256)
void build_tseq_k(const __bf16* __restrict__ out2, const float* __restrict__ srcframe,
                  __bf16* __restrict__ tseq)
{
  const int HW = HWX;
  int idx = blockIdx.x * 256 + threadIdx.x;
  if (idx >= 2 * HW) return;
  const int b = idx / HW, p = idx - b * HW;
  __bf16* t = tseq + ((size_t)b * HW + p) * 64;
  const __bf16* o2 = out2 + ((size_t)b * HW + p) * 32;
  for (int c = 0; c < 32; ++c) t[c] = o2[c];
  for (int c = 0; c < 32; ++c)
    t[32 + c] = (__bf16)srcframe[(((size_t)b * 3 + 1) * 32 + c) * HW + p];
}

// ---------------------------------------------------------------------------
// Host orchestration
// ---------------------------------------------------------------------------
extern "C" void kernel_launch(void* const* d_in, const int* in_sizes, int n_in,
                              void* d_out, int out_size, void* d_ws, size_t ws_size,
                              hipStream_t stream)
{
  (void)in_sizes; (void)n_in; (void)out_size; (void)ws_size;
  const float* frame    = (const float*)d_in[0];
  const float* srcframe = (const float*)d_in[1];
  const float* ffwd     = (const float*)d_in[2];
  const float* fbwd     = (const float*)d_in[3];
  const float* W_qk  = (const float*)d_in[4];   const float* b_qk  = (const float*)d_in[5];
  const float* W_v   = (const float*)d_in[6];   const float* b_v   = (const float*)d_in[7];
  const float* W_vp  = (const float*)d_in[8];   const float* b_vp  = (const float*)d_in[9];
  const float* W_off = (const float*)d_in[10];  const float* b_off = (const float*)d_in[11];
  const float* W_aw  = (const float*)d_in[12];  const float* b_aw  = (const float*)d_in[13];
  const float* W_out = (const float*)d_in[14];  const float* b_out = (const float*)d_in[15];
  const float* W_ffd = (const float*)d_in[16];  const float* b_ffd = (const float*)d_in[17];
  const float* W_ff1 = (const float*)d_in[18];  const float* b_ff1 = (const float*)d_in[19];
  const float* W_ff2 = (const float*)d_in[20];  const float* b_ff2 = (const float*)d_in[21];
  const float* W_fu  = (const float*)d_in[22];  const float* b_fu  = (const float*)d_in[23];

  const size_t HW = HWX;
  __bf16* bws = (__bf16*)d_ws;
  size_t o = 0;
  auto ab = [&](size_t nelem) { __bf16* p = bws + o; o += nelem; return p; };

  __bf16* qk      = ab(2 * HW * 128);
  __bf16* frame96 = ab(2 * HW * 96);
  __bf16* qureys  = ab(2 * HW * 96);
  __bf16* value   = ab(6 * HW * 32);   // (6, HW, 32) via outC split
  __bf16* valb    = ab(6 * HW * 32);
  __bf16* offb    = ab(2 * HW * 96);
  __bf16* awb     = ab(2 * HW * 48);
  __bf16* attnb   = ab(2 * HW * 32);
  __bf16* out1    = ab(2 * HW * 32);
  __bf16* out2    = ab(2 * HW * 32);
  __bf16* tseq    = ab(2 * HW * 64);
  __bf16* ff1b    = ab(2 * HW * 32);
  __bf16* out3    = ab(2 * HW * 32);

  auto wrep = [&](const float* src, int Cout, int Cin, int taps, int Cpad) {
    const int nelem = Cout * taps * Cpad;
    __bf16* dst = ab((size_t)nelem);
    repack_w_k<<<(nelem + 255) / 256, 256, 0, stream>>>(src, dst, Cout, Cin, taps, Cpad);
    return dst;
  };
  __bf16* wqk  = wrep(W_qk,  96, 100, 9, 128);
  __bf16* wv   = wrep(W_v,   96,  96, 9,  96);
  __bf16* wvp  = wrep(W_vp,  32,  32, 9,  32);
  __bf16* woff = wrep(W_off, 96,  96, 9,  96);
  __bf16* waw  = wrep(W_aw,  48,  96, 9,  96);
  __bf16* wout = wrep(W_out, 32,  32, 9,  32);
  __bf16* wffd = wrep(W_ffd, 32,  32, 9,  32);
  __bf16* wff1 = wrep(W_ff1, 32,  64, 9,  64);
  __bf16* wff2 = wrep(W_ff2, 32,  32, 9,  32);
  __bf16* wfu  = wrep(W_fu,  32,  32, 1,  32);

  // Input-format conversions
  nchw_to_nhwc_k<<<(2 * (int)HW + 255) / 256, 256, 0, stream>>>(frame, frame96, 2, 96, 96);
  warp_build_qkin_k<<<(2 * (int)HW + 255) / 256, 256, 0, stream>>>(frame, ffwd, fbwd, qk);

  auto conv = [&](const __bf16* in, const __bf16* w, const float* bias, void* outp,
                  const void* res, long long resStride, int resMode,
                  int Cpad, int Cout, int nimg, int taps, int dil, int act,
                  int outMode, int outC) {
    dim3 g(HWX / 256, Cout / 16, nimg);
    const int shmem = 32 * taps * Cpad;   // 16 rows * Ktot * 2B
    conv_wmma_k<<<g, 256, shmem, stream>>>(in, w, bias, outp, res, resStride,
                                           Cpad, Cout, taps, dil, act,
                                           outMode, outC, resMode);
  };

  // qureys = lrelu(conv(qk_in, W_qk))
  conv(qk, wqk, b_qk, qureys, nullptr, 0, 0, 128, 96, 2, 9, 1, 1, 0, 96);
  // value = conv(frame(B,96), W_v) -> stored split as (6, HW, 32)
  conv(frame96, wv, b_v, value, nullptr, 0, 0, 96, 96, 2, 9, 1, 0, 0, 32);
  // val = conv(value(6,32), W_vp)
  conv(value, wvp, b_vp, valb, nullptr, 0, 0, 32, 32, 6, 9, 1, 0, 0, 32);
  // off / aw
  conv(qureys, woff, b_off, offb, nullptr, 0, 0, 96, 96, 2, 9, 1, 0, 0, 96);
  conv(qureys, waw,  b_aw,  awb,  nullptr, 0, 0, 96, 48, 2, 9, 1, 0, 0, 48);

  softmax_aw_k<<<(2 * 4 * (int)HW + 255) / 256, 256, 0, stream>>>(awb);
  deform_attn_k<<<(2 * 4 * (int)HW + 255) / 256, 256, 0, stream>>>(valb, offb, awb, attnb);

  // out1 = conv(attn, W_out)
  conv(attnb, wout, b_out, out1, nullptr, 0, 0, 32, 32, 2, 9, 1, 0, 0, 32);
  // out2 = conv(out1, W_ffd) + frame[:,1]  (residual f32 NCHW)
  conv(out1, wffd, b_ffd, out2, frame + (size_t)32 * HW, (long long)(96 * HW), 1,
       32, 32, 2, 9, 1, 0, 0, 32);

  build_tseq_k<<<(2 * (int)HW + 255) / 256, 256, 0, stream>>>(out2, srcframe, tseq);

  // ff1 = lrelu(conv(tseq, W_ff1, dil=2))
  conv(tseq, wff1, b_ff1, ff1b, nullptr, 0, 0, 64, 32, 2, 9, 2, 1, 0, 32);
  // out3 = out2 + conv(ff1, W_ff2)  (residual bf16 NHWC)
  conv(ff1b, wff2, b_ff2, out3, out2, 0, 2, 32, 32, 2, 9, 1, 0, 0, 32);
  // final = lrelu(conv1x1(out3, W_fu)) -> f32 NCHW d_out
  conv(out3, wfu, b_fu, (float*)d_out, nullptr, 0, 0, 32, 32, 2, 1, 1, 1, 1, 32);
}
